// ExpertsImage_16896401343010
// MI455X (gfx1250) — compile-verified, matrix operand order checked by reading, not run on hardware
//
#include <hip/hip_runtime.h>

// ---------------------------------------------------------------------------
// MoE top-2 on gfx1250 (wave32):
//   1) moe_init  : zero per-expert counters
//   2) moe_gate  : softmax+top2 per token (one wave/token), build expert lists
//   3) moe_prep  : expert_w f32 -> K-tiled N-major bf16 hi/lo planes (ws)
//   4) moe_expert_gemm_pre : grouped GEMM, 64x128 tile/WG, DOUBLE-BUFFERED
//        async-to-LDS B staging (ASYNCcnt pipelined: s_wait_asynccnt <= 4),
//        12x v_wmma_f32_16x16x32_bf16 per K-step per wave (split-bf16 fp32).
//   (fallback: moe_expert_gemm_fly converts weights on the fly if ws small)
// ---------------------------------------------------------------------------

typedef __attribute__((ext_vector_type(16))) __bf16        v16bf;
typedef __attribute__((ext_vector_type(8)))  float         v8f;
typedef __attribute__((ext_vector_type(4)))  unsigned int  v4u;

union Frag { v16bf v; v4u q[2]; };

constexpr int B_  = 4;
constexpr int S_  = 2048;
constexpr int E_  = 1024;
constexpr int H_  = 2048;
constexpr int NE_ = 10;
constexpr int T_  = B_ * S_;     // 8192 tokens
constexpr int A_  = 2 * T_;      // 16384 (token,slot) assignments
constexpr int MAXPE = T_;        // worst-case assignments per expert
constexpr int KT_  = E_ / 32;    // 32 K-tiles per expert

__device__ __forceinline__ unsigned short f2bf(float f) {
    unsigned int u = __float_as_uint(f);
    u += 0x7FFFu + ((u >> 16) & 1u);          // round-to-nearest-even
    return (unsigned short)(u >> 16);
}
__device__ __forceinline__ float bf2f(unsigned short h) {
    return __uint_as_float(((unsigned int)h) << 16);
}
__device__ __forceinline__ unsigned int lds_off32(const void* p) {
    // generic LDS pointer: high bits = shared aperture, low 32 = LDS offset
    return (unsigned int)(unsigned long long)p;
}
__device__ __forceinline__ void async_copy_b128(unsigned int lds_off,
                                                const void* gsrc) {
    asm volatile("global_load_async_to_lds_b128 %0, %1, off"
                 :: "v"(lds_off), "v"(gsrc) : "memory");
}
__device__ __forceinline__ void wait_async_le4() {
    asm volatile("s_wait_asynccnt 0x4" ::: "memory");
}
__device__ __forceinline__ void wait_async0() {
    asm volatile("s_wait_asynccnt 0x0" ::: "memory");
}

// --------------------------- kernel 1: init --------------------------------
__global__ void moe_init(int* counts) {
    if (threadIdx.x < 16) counts[threadIdx.x] = 0;
}

// --------------------------- kernel 2: gating ------------------------------
__global__ void __launch_bounds__(256) moe_gate(
    const float* __restrict__ x,       // [T,E]
    const float* __restrict__ gw,      // [E,NE]
    const float* __restrict__ gb,      // [NE]
    float*       __restrict__ out_w,   // d_out: first A_ floats
    int*         __restrict__ counts,
    int*         __restrict__ lists) {
    const int lane  = threadIdx.x & 31;
    const int token = blockIdx.x * 8 + (threadIdx.x >> 5);

    const float* xr = x + (size_t)token * E_;
    float acc[NE_];
#pragma unroll
    for (int n = 0; n < NE_; ++n) acc[n] = 0.f;

    for (int e = lane; e < E_; e += 32) {
        const float xv = xr[e];
        const float* g = gw + e * NE_;
#pragma unroll
        for (int n = 0; n < NE_; ++n) acc[n] += xv * g[n];
    }
#pragma unroll
    for (int off = 16; off > 0; off >>= 1)
#pragma unroll
        for (int n = 0; n < NE_; ++n) acc[n] += __shfl_xor(acc[n], off, 32);

    if (lane == 0) {
        float l[NE_];
#pragma unroll
        for (int n = 0; n < NE_; ++n) l[n] = acc[n] + gb[n];
        int i0 = 0; float v0 = l[0];
#pragma unroll
        for (int n = 1; n < NE_; ++n) if (l[n] > v0) { v0 = l[n]; i0 = n; }
        int i1 = (i0 == 0) ? 1 : 0; float v1 = l[i1];
#pragma unroll
        for (int n = 0; n < NE_; ++n)
            if (n != i0 && l[n] > v1) { v1 = l[n]; i1 = n; }
        float s = 0.f;
#pragma unroll
        for (int n = 0; n < NE_; ++n) s += __expf(l[n] - v0);
        const float inv = 1.f / s;
        out_w[token * 2 + 0] = inv;
        out_w[token * 2 + 1] = __expf(v1 - v0) * inv;

        const int p0 = atomicAdd(&counts[i0], 1);
        lists[i0 * MAXPE + p0] = token * 2 + 0;
        const int p1 = atomicAdd(&counts[i1], 1);
        lists[i1 * MAXPE + p1] = token * 2 + 1;
    }
}

// ---------------- kernel 3: pre-convert expert_w to bf16 hi/lo -------------
// plane layout: [e][kt][h][kk] , kk in [0,32) contiguous (N-major K-tiles)
__global__ void __launch_bounds__(256) moe_prep(
    const float*    __restrict__ ew,   // [NE,E,H]
    unsigned short* __restrict__ bh,
    unsigned short* __restrict__ bl) {
    const int e  = blockIdx.z;
    const int kt = blockIdx.y;
    const int h0 = blockIdx.x * 128;
    const int tid = threadIdx.x;

    __shared__ __align__(16) unsigned short Th[4096];  // [128][32]
    __shared__ __align__(16) unsigned short Tl[4096];

    const float* src = ew + ((size_t)e * E_ + kt * 32) * H_ + h0;
#pragma unroll
    for (int i = 0; i < 4; ++i) {
        const int q  = tid + i * 256;
        const int kk = q >> 5;
        const int nn = (q & 31) * 4;
        const float4 w4 = *reinterpret_cast<const float4*>(
            src + (size_t)kk * H_ + nn);
        const float wv[4] = {w4.x, w4.y, w4.z, w4.w};
#pragma unroll
        for (int j = 0; j < 4; ++j) {
            const unsigned short h = f2bf(wv[j]);
            Th[(nn + j) * 32 + kk] = h;
            Tl[(nn + j) * 32 + kk] = f2bf(wv[j] - bf2f(h));
        }
    }
    __syncthreads();
    const size_t dst = ((size_t)(e * KT_ + kt) * H_ + h0) * 32;  // ushort idx
    const v4u* th = reinterpret_cast<const v4u*>(Th);
    const v4u* tl = reinterpret_cast<const v4u*>(Tl);
    v4u* oh = reinterpret_cast<v4u*>(bh + dst);
    v4u* ol = reinterpret_cast<v4u*>(bl + dst);
    oh[tid * 2 + 0] = th[tid * 2 + 0];
    oh[tid * 2 + 1] = th[tid * 2 + 1];
    ol[tid * 2 + 0] = tl[tid * 2 + 0];
    ol[tid * 2 + 1] = tl[tid * 2 + 1];
}

// -------- kernel 4: grouped GEMM, pre-converted weights, async B stage -----
// WG = 256 thr = 8 waves; tile = 64 assignments x 128 H; 4 acc per wave.
// Double-buffered LDS: next K-tile's async copies overlap current compute.
__global__ void __launch_bounds__(256, 2) moe_expert_gemm_pre(
    const float*          __restrict__ x,
    const unsigned short* __restrict__ bh,
    const unsigned short* __restrict__ bl,
    const float*          __restrict__ eb,
    const int*            __restrict__ counts,
    const int*            __restrict__ lists,
    float*                __restrict__ out) {
    const int e   = blockIdx.z;
    const int cnt = counts[e];
    const int m0  = blockIdx.y * 64;
    if (m0 >= cnt) return;                       // wave-uniform exit
    const int n0   = blockIdx.x * 128;
    const int tid  = threadIdx.x;
    const int lane = tid & 31;
    const int wave = tid >> 5;

    __shared__ int s_asg[64];
    __shared__ __align__(16) unsigned short Ah[2][64 * 32];
    __shared__ __align__(16) unsigned short Al[2][64 * 32];
    __shared__ __align__(16) unsigned short Bh[2][4096];   // [128][32] N-major
    __shared__ __align__(16) unsigned short Bl[2][4096];

    if (tid < 64) {
        const int mi = m0 + tid;
        s_asg[tid] = (mi < cnt) ? lists[e * MAXPE + mi] : -1;
    }
    __syncthreads();

    // A staging: thread owns row arow, 8 consecutive K (two float4)
    const int arow = tid >> 2;          // 0..63
    const int ac0  = (tid & 3) * 8;     // 0,8,16,24
    const int aasg = s_asg[arow];
    const float* asrc =
        (aasg >= 0) ? (x + (size_t)(aasg >> 1) * E_ + ac0) : nullptr;

    // B async staging: per-kt slab of 128x32 ushorts (8 KB contiguous)
    const unsigned short* bhs = bh + ((size_t)(e * KT_) * H_ + n0) * 32;
    const unsigned short* bls = bl + ((size_t)(e * KT_) * H_ + n0) * 32;
    const unsigned int ldsBh[2] = {lds_off32(&Bh[0][0]), lds_off32(&Bh[1][0])};
    const unsigned int ldsBl[2] = {lds_off32(&Bl[0][0]), lds_off32(&Bl[1][0])};

    v8f acc0 = {}, acc1 = {}, acc2 = {}, acc3 = {};
    const int kb = (lane >> 4) * 8;          // K-base (16-bit operand layout)
    const int bn = wave * 16 + (lane & 15);  // B column within 128-wide tile
    const int am = lane & 15;

    // ---- stage K-tile ktn into buffer b (async B copy + A split stores) ----
    auto stage = [&](int ktn, int b) {
        const unsigned short* sh = bhs + (size_t)ktn * H_ * 32;
        const unsigned short* sl = bls + (size_t)ktn * H_ * 32;
        async_copy_b128(ldsBh[b] + tid * 16, (const char*)sh + tid * 16);
        async_copy_b128(ldsBh[b] + tid * 16 + 4096,
                        (const char*)sh + tid * 16 + 4096);
        async_copy_b128(ldsBl[b] + tid * 16, (const char*)sl + tid * 16);
        async_copy_b128(ldsBl[b] + tid * 16 + 4096,
                        (const char*)sl + tid * 16 + 4096);

        float4 xa = make_float4(0.f, 0.f, 0.f, 0.f), xb = xa;
        if (asrc) {
            xa = *reinterpret_cast<const float4*>(asrc + ktn * 32);
            xb = *reinterpret_cast<const float4*>(asrc + ktn * 32 + 4);
        }
        const float av[8] = {xa.x, xa.y, xa.z, xa.w, xb.x, xb.y, xb.z, xb.w};
#pragma unroll
        for (int j = 0; j < 8; ++j) {
            const unsigned short h = f2bf(av[j]);
            Ah[b][arow * 32 + ac0 + j] = h;
            Al[b][arow * 32 + ac0 + j] = f2bf(av[j] - bf2f(h));
        }
    };

    // ---- consume buffer b: fragments + 12 WMMAs --------------------------
    auto compute = [&](int b) {
        Frag bhf, blf;
        bhf.q[0] = *reinterpret_cast<const v4u*>(&Bh[b][bn * 32 + kb]);
        bhf.q[1] = *reinterpret_cast<const v4u*>(&Bh[b][bn * 32 + kb + 16]);
        blf.q[0] = *reinterpret_cast<const v4u*>(&Bl[b][bn * 32 + kb]);
        blf.q[1] = *reinterpret_cast<const v4u*>(&Bl[b][bn * 32 + kb + 16]);
#pragma unroll
        for (int ms = 0; ms < 4; ++ms) {
            Frag ah, al;
            const int r = (ms * 16 + am) * 32;
            ah.q[0] = *reinterpret_cast<const v4u*>(&Ah[b][r + kb]);
            ah.q[1] = *reinterpret_cast<const v4u*>(&Ah[b][r + kb + 16]);
            al.q[0] = *reinterpret_cast<const v4u*>(&Al[b][r + kb]);
            al.q[1] = *reinterpret_cast<const v4u*>(&Al[b][r + kb + 16]);
            v8f a = (ms == 0) ? acc0 : (ms == 1) ? acc1 : (ms == 2) ? acc2 : acc3;
            a = __builtin_amdgcn_wmma_f32_16x16x32_bf16(false, ah.v, false, bhf.v,
                                                        (short)0, a, false, false);
            a = __builtin_amdgcn_wmma_f32_16x16x32_bf16(false, al.v, false, bhf.v,
                                                        (short)0, a, false, false);
            a = __builtin_amdgcn_wmma_f32_16x16x32_bf16(false, ah.v, false, blf.v,
                                                        (short)0, a, false, false);
            if (ms == 0) acc0 = a; else if (ms == 1) acc1 = a;
            else if (ms == 2) acc2 = a; else acc3 = a;
        }
    };

    // ---- software pipeline over K --------------------------------------
    int buf = 0;
    stage(0, 0);
    for (int kt = 0; kt < KT_ - 1; ++kt) {
        stage(kt + 1, buf ^ 1);   // next tile in flight (other buffer)
        wait_async_le4();         // in-order: current tile's 4 copies done
        __syncthreads();
        compute(buf);
        __syncthreads();          // all reads of buf latched before reuse
        buf ^= 1;
    }
    wait_async0();
    __syncthreads();
    compute(buf);

    // ---- bias + store ----
    const int gn = n0 + wave * 16 + (lane & 15);
    const float bias = eb[e * H_ + gn];
#pragma unroll
    for (int ms = 0; ms < 4; ++ms) {
        const v8f a = (ms == 0) ? acc0 : (ms == 1) ? acc1 : (ms == 2) ? acc2 : acc3;
#pragma unroll
        for (int r = 0; r < 8; ++r) {
            const int m  = ms * 16 + r + ((lane >> 4) << 3);
            const int mi = m0 + m;
            if (mi < cnt) {
                const int asg = s_asg[m];
                out[(size_t)A_ + (size_t)asg * H_ + gn] = a[r] + bias;
            }
        }
    }
}

// ------------- fallback GEMM: on-the-fly conversion (small ws) -------------
__global__ void __launch_bounds__(256, 2) moe_expert_gemm_fly(
    const float* __restrict__ x,
    const float* __restrict__ ew,
    const float* __restrict__ eb,
    const int*   __restrict__ counts,
    const int*   __restrict__ lists,
    float*       __restrict__ out) {
    const int e   = blockIdx.z;
    const int cnt = counts[e];
    const int m0  = blockIdx.y * 16;
    if (m0 >= cnt) return;
    const int n0   = blockIdx.x * 128;
    const int tid  = threadIdx.x;
    const int lane = tid & 31;
    const int wave = tid >> 5;

    __shared__ int s_asg[16];
    __shared__ __align__(16) unsigned short Ah[16][32];
    __shared__ __align__(16) unsigned short Al[16][32];
    __shared__ __align__(16) unsigned short Bh[128][32];
    __shared__ __align__(16) unsigned short Bl[128][32];

    if (tid < 16) {
        const int mi = m0 + tid;
        s_asg[tid] = (mi < cnt) ? lists[e * MAXPE + mi] : -1;
    }
    __syncthreads();

    const int arow  = tid >> 4;
    const int acol  = (tid & 15) * 2;
    const int a_asg = s_asg[arow];
    const float* a_src =
        (a_asg >= 0) ? (x + (size_t)(a_asg >> 1) * E_ + acol) : nullptr;
    const float* ew_e = ew + (size_t)e * E_ * H_;

    v8f acc = {};
    const int kb = (lane >> 4) * 8;
    const int am = lane & 15;
    const int bn = wave * 16 + (lane & 15);

    for (int k0 = 0; k0 < E_; k0 += 32) {
        {
            float ax = 0.f, ay = 0.f;
            if (a_src) {
                const float2 xv = *reinterpret_cast<const float2*>(a_src + k0);
                ax = xv.x; ay = xv.y;
            }
            const unsigned short hx = f2bf(ax);
            const unsigned short hy = f2bf(ay);
            Ah[arow][acol + 0] = hx;  Al[arow][acol + 0] = f2bf(ax - bf2f(hx));
            Ah[arow][acol + 1] = hy;  Al[arow][acol + 1] = f2bf(ay - bf2f(hy));
        }
#pragma unroll
        for (int i = 0; i < 4; ++i) {
            const int q  = tid + i * 256;
            const int kk = q >> 5;
            const int nn = (q & 31) * 4;
            const float4 w4 = *reinterpret_cast<const float4*>(
                ew_e + (size_t)(k0 + kk) * H_ + n0 + nn);
            const float wv[4] = {w4.x, w4.y, w4.z, w4.w};
#pragma unroll
            for (int j = 0; j < 4; ++j) {
                const unsigned short h = f2bf(wv[j]);
                Bh[nn + j][kk] = h;
                Bl[nn + j][kk] = f2bf(wv[j] - bf2f(h));
            }
        }
        __syncthreads();

        Frag ah, al, bhf, blf;
        ah.q[0] = *reinterpret_cast<const v4u*>(&Ah[am][kb]);
        ah.q[1] = *reinterpret_cast<const v4u*>(&Ah[am][kb + 16]);
        al.q[0] = *reinterpret_cast<const v4u*>(&Al[am][kb]);
        al.q[1] = *reinterpret_cast<const v4u*>(&Al[am][kb + 16]);
        bhf.q[0] = *reinterpret_cast<const v4u*>(&Bh[bn][kb]);
        bhf.q[1] = *reinterpret_cast<const v4u*>(&Bh[bn][kb + 16]);
        blf.q[0] = *reinterpret_cast<const v4u*>(&Bl[bn][kb]);
        blf.q[1] = *reinterpret_cast<const v4u*>(&Bl[bn][kb + 16]);

        acc = __builtin_amdgcn_wmma_f32_16x16x32_bf16(false, ah.v, false, bhf.v,
                                                      (short)0, acc, false, false);
        acc = __builtin_amdgcn_wmma_f32_16x16x32_bf16(false, al.v, false, bhf.v,
                                                      (short)0, acc, false, false);
        acc = __builtin_amdgcn_wmma_f32_16x16x32_bf16(false, ah.v, false, blf.v,
                                                      (short)0, acc, false, false);
        __syncthreads();
    }

    const int gn = n0 + wave * 16 + (lane & 15);
    const float bias = eb[e * H_ + gn];
#pragma unroll
    for (int r = 0; r < 8; ++r) {
        const int m  = r + ((lane >> 4) << 3);
        const int mi = m0 + m;
        if (mi < cnt) {
            const int asg = s_asg[m];
            out[(size_t)A_ + (size_t)asg * H_ + gn] = acc[r] + bias;
        }
    }
}

// ---------------------------------------------------------------------------
extern "C" void kernel_launch(void* const* d_in, const int* in_sizes, int n_in,
                              void* d_out, int out_size, void* d_ws, size_t ws_size,
                              hipStream_t stream) {
    (void)in_sizes; (void)n_in; (void)out_size;
    const float* x  = (const float*)d_in[0];
    const float* gw = (const float*)d_in[1];
    const float* gb = (const float*)d_in[2];
    const float* ew = (const float*)d_in[3];
    const float* eb = (const float*)d_in[4];
    float* out = (float*)d_out;

    int* counts = (int*)d_ws;       // 16 ints
    int* lists  = counts + 16;      // NE*MAXPE ints (~320 KB)

    const size_t off_planes = 327936;                        // 256B-aligned
    const size_t plane_elts = (size_t)NE_ * E_ * H_;         // 20,971,520
    const size_t need = off_planes + 2 * plane_elts * sizeof(unsigned short);

    moe_init<<<1, 32, 0, stream>>>(counts);
    moe_gate<<<T_ / 8, 256, 0, stream>>>(x, gw, gb, out, counts, lists);

    if (ws_size >= need) {
        unsigned short* bh = (unsigned short*)((char*)d_ws + off_planes);
        unsigned short* bl = bh + plane_elts;
        dim3 gp(H_ / 128, KT_, NE_);
        moe_prep<<<gp, 256, 0, stream>>>(ew, bh, bl);
        dim3 gg(H_ / 128, (T_ + 63) / 64, NE_);
        moe_expert_gemm_pre<<<gg, 256, 0, stream>>>(x, bh, bl, eb, counts,
                                                    lists, out);
    } else {
        dim3 gg(H_ / 128, (T_ + 15) / 16, NE_);
        moe_expert_gemm_fly<<<gg, 256, 0, stream>>>(x, ew, eb, counts, lists,
                                                    out);
    }
}
